// MultiHeadLatentAttention_17892833755770
// MI455X (gfx1250) — matrix operand order repaired and checked by reading, not compiled, not run on hardware
//
#include <hip/hip_runtime.h>

// ---------------------------------------------------------------------------
// MLA forward for MI455X (gfx1250, wave32, WMMA).
// All matmuls run on v_wmma_f32_16x16x32_bf16 (f32 accumulate).
// GEMMs templated on (N,K): immediate address math, b128 staging, software
// pipelining of global loads, and double-buffered LDS fragment loads so the
// WMMA stream does not wait on dscnt==0 every step.
// ---------------------------------------------------------------------------

#define B_  2
#define S_  2048
#define D_  2048
#define H_  16
#define DN_ 128
#define DR_ 64
#define DV_ 128
#define CKV_ 512
#define CQ_ 1536
#define MTOT (B_ * S_)          // 4096 rows for all projection GEMMs
#define DQK (DN_ + DR_)         // 192

typedef __attribute__((ext_vector_type(16))) __bf16 v16bf;
typedef __attribute__((ext_vector_type(8)))  float  v8f;
typedef __attribute__((ext_vector_type(4)))  __bf16 bf16x4;

__device__ __forceinline__ bf16x4 cvt4(float4 v) {
  bf16x4 h;
  h.x = (__bf16)v.x; h.y = (__bf16)v.y; h.z = (__bf16)v.z; h.w = (__bf16)v.w;
  return h;
}

// ---------------------------------------------------------------------------
// GEMM: C[M,N] = A[M,K] @ B[K,N], f32 in/out, bf16 WMMA inside.
// Block tile 128(M) x BN(N), K-step 32, 256 threads = 8 waves.
// ---------------------------------------------------------------------------
#define GBM 128
#define GKB 32

template<int N, int K>
__global__ __launch_bounds__(256) void gemm_bf16_wmma(
    const float* __restrict__ A, const float* __restrict__ Bw,
    float* __restrict__ C)
{
  constexpr int BN  = (N >= 128) ? 128 : 64;
  constexpr int NT  = BN / 16;    // accumulators per wave
  constexpr int NB4 = BN / 32;    // B float4 loads per thread per tile
  constexpr int BNV = BN / 4;     // float4 per B tile row

  __shared__ __bf16 As[GBM][GKB + 8];     // [m][k]
  __shared__ __bf16 Bst[BN][GKB + 8];     // [n][k] transposed

  const int tid  = threadIdx.x;
  const int wave = tid >> 5;
  const int lane = tid & 31;
  const int hi   = lane >> 4;
  const int nl   = lane & 15;
  const int bm   = blockIdx.x * GBM;
  const int bn   = blockIdx.y * BN;
  const int m0   = wave * 16;

  v8f acc[NT];
  const v8f vzero = {0.f,0.f,0.f,0.f,0.f,0.f,0.f,0.f};
#pragma unroll
  for (int t = 0; t < NT; ++t) acc[t] = vzero;

  float4 aR[4], bR[NB4];

  auto gload = [&](int k0) {
#pragma unroll
    for (int i = 0; i < 4; ++i) {               // A: 128x32 tile, 4 float4/thr
      int idx = i * 256 + tid;
      int r = idx >> 3, sg = idx & 7;
      aR[i] = *(const float4*)(A + (size_t)(bm + r) * K + k0 + sg * 4);
    }
#pragma unroll
    for (int i = 0; i < NB4; ++i) {             // B: 32xBN tile
      int idx = i * 256 + tid;
      int kr = idx / BNV, sg = idx % BNV;
      bR[i] = *(const float4*)(Bw + (size_t)(k0 + kr) * N + bn + sg * 4);
    }
  };
  auto lstore = [&]() {
#pragma unroll
    for (int i = 0; i < 4; ++i) {
      int idx = i * 256 + tid;
      int r = idx >> 3, sg = idx & 7;
      *(bf16x4*)&As[r][sg * 4] = cvt4(aR[i]);   // ds_store_b64
    }
#pragma unroll
    for (int i = 0; i < NB4; ++i) {             // transpose scatter (b16 stores)
      int idx = i * 256 + tid;
      int kr = idx / BNV, sg = idx % BNV;
      Bst[sg * 4 + 0][kr] = (__bf16)bR[i].x;
      Bst[sg * 4 + 1][kr] = (__bf16)bR[i].y;
      Bst[sg * 4 + 2][kr] = (__bf16)bR[i].z;
      Bst[sg * 4 + 3][kr] = (__bf16)bR[i].w;
    }
  };

  const int kb8 = hi << 3;
  const int ks  = hi << 4;

  auto ldB = [&](int t) {
    v16bf b;
    int nc = t * 16 + nl;
#pragma unroll
    for (int i = 0; i < 16; ++i) b[i] = Bst[nc][kb8 + (i & 7) + ((i >> 3) << 4)];
    return b;
  };

  gload(0);
  for (int k0 = 0; k0 < K; k0 += GKB) {
    lstore();
    __syncthreads();
    if (k0 + GKB < K) gload(k0 + GKB);          // prefetch next tile (hidden)

    // ---- A fragment (ISA 16-bit A 16x32 layout): 2 contiguous 16B chunks ----
    v16bf af;
#pragma unroll
    for (int v = 0; v < 8; ++v) {
      int kk = (v < 4) ? (kb8 + 2 * v) : (16 + kb8 + 2 * (v - 4));
      af[2 * v]     = As[m0 + nl][kk];
      af[2 * v + 1] = As[m0 + nl][kk + 1];
    }
    // ---- double-buffered B fragments so WMMA never waits dscnt==0 ----
    v16bf bcur = ldB(0);
#pragma unroll
    for (int t = 0; t < NT; ++t) {
      v16bf bnxt;
      if (t + 1 < NT) bnxt = ldB(t + 1);
      acc[t] = __builtin_amdgcn_wmma_f32_16x16x32_bf16(
          false, af, false, bcur, (short)0, acc[t], false, false);
      if (t + 1 < NT) bcur = bnxt;
    }
    __syncthreads();
  }

  // ---- write C: immediate-offset stores off one base pointer ----
  float* cp = C + (size_t)(bm + m0 + 8 * hi) * N + bn + nl;
#pragma unroll
  for (int t = 0; t < NT; ++t)
#pragma unroll
    for (int r = 0; r < 8; ++r)
      cp[r * N + t * 16] = acc[t][r];
}

// ---------------------------------------------------------------------------
// In-place RMSNorm over rows of length N (one workgroup per row).
// ---------------------------------------------------------------------------
__global__ __launch_bounds__(256) void rmsnorm_inplace(
    float* __restrict__ X, const float* __restrict__ g, int N)
{
  float* x = X + (size_t)blockIdx.x * N;
  __shared__ float red[8];
  __shared__ float s_scale;

  float ss = 0.f;
  for (int i = threadIdx.x; i < N; i += 256) { float v = x[i]; ss += v * v; }
#pragma unroll
  for (int off = 16; off > 0; off >>= 1) ss += __shfl_xor(ss, off, 32);
  if ((threadIdx.x & 31) == 0) red[threadIdx.x >> 5] = ss;
  __syncthreads();
  if (threadIdx.x == 0) {
    float t = 0.f;
#pragma unroll
    for (int i = 0; i < 8; ++i) t += red[i];
    s_scale = rsqrtf(t / (float)N + 1e-6f);
  }
  __syncthreads();
  float sc = s_scale;
  for (int i = threadIdx.x; i < N; i += 256) x[i] = x[i] * sc * g[i];
}

// ---------------------------------------------------------------------------
// RoPE helper: cos/sin for position s, rope-dim d in [0,64)
// ---------------------------------------------------------------------------
__device__ __forceinline__ float2 rope_cs(int s, int d) {
  int j = d & 31;
  float invf = __expf(-(float)j * (1.0f / 32.0f) * 9.2103403719761836f); // ln(1e4)
  float ang = (float)s * invf;
  return make_float2(__cosf(ang), __sinf(ang));
}

// ---------------------------------------------------------------------------
// Pack kernels: f32 GEMM outputs -> bf16 [B,H,S,d] tensors (RoPE fused).
// ---------------------------------------------------------------------------
__global__ __launch_bounds__(256) void pack_q(
    const float* __restrict__ qn, const float* __restrict__ qr,
    __bf16* __restrict__ qp)
{
  int idx = blockIdx.x * 256 + threadIdx.x;
  int d = idx % DQK;
  int s = (idx / DQK) % S_;
  int h = (idx / (DQK * S_)) % H_;
  int b = idx / (DQK * S_ * H_);
  size_t bs = (size_t)b * S_ + s;
  float val;
  if (d < DN_) {
    val = qn[bs * (H_ * DN_) + h * DN_ + d];
  } else {
    int dr = d - DN_;
    const float* base = qr + bs * (H_ * DR_) + h * DR_;
    float xv = base[dr];
    float rot = (dr < 32) ? -base[dr + 32] : base[dr - 32];
    float2 cs = rope_cs(s, dr);
    val = xv * cs.x + rot * cs.y;
  }
  qp[idx] = (__bf16)val;
}

__global__ __launch_bounds__(256) void pack_k(
    const float* __restrict__ kn, const float* __restrict__ kr,
    __bf16* __restrict__ kp)
{
  int idx = blockIdx.x * 256 + threadIdx.x;
  int d = idx % DQK;
  int s = (idx / DQK) % S_;
  int h = (idx / (DQK * S_)) % H_;
  int b = idx / (DQK * S_ * H_);
  size_t bs = (size_t)b * S_ + s;
  float val;
  if (d < DN_) {
    val = kn[bs * (H_ * DN_) + h * DN_ + d];
  } else {
    int dr = d - DN_;
    const float* base = kr + bs * DR_;
    float xv = base[dr];
    float rot = (dr < 32) ? -base[dr + 32] : base[dr - 32];
    float2 cs = rope_cs(s, dr);
    val = xv * cs.x + rot * cs.y;
  }
  kp[idx] = (__bf16)val;
}

__global__ __launch_bounds__(256) void pack_v(
    const float* __restrict__ vf, __bf16* __restrict__ vp)
{
  int idx = blockIdx.x * 256 + threadIdx.x;
  int d = idx % DV_;
  int s = (idx / DV_) % S_;
  int h = (idx / (DV_ * S_)) % H_;
  int b = idx / (DV_ * S_ * H_);
  size_t bs = (size_t)b * S_ + s;
  vp[idx] = (__bf16)vf[bs * (H_ * DV_) + h * DV_ + d];
}

// ---------------------------------------------------------------------------
// Flash attention with WMMA.  Grid: (S/64, B*H); block 128 threads (4 waves).
// ---------------------------------------------------------------------------
#define FA_BM 64
#define FA_BK 64

__global__ __launch_bounds__(128) void fa_kernel(
    const __bf16* __restrict__ Q,   // [B*H, S, DQK]
    const __bf16* __restrict__ Kp,  // [B*H, S, DQK]
    const __bf16* __restrict__ Vp,  // [B*H, S, DV]
    float* __restrict__ O)          // [B, S, H*DV]
{
  const int bh = blockIdx.y;
  const int b  = bh / H_;
  const int h  = bh % H_;
  const int q0 = blockIdx.x * FA_BM;
  const int tid  = threadIdx.x;
  const int wave = tid >> 5;
  const int lane = tid & 31;
  const int hi   = lane >> 4;
  const int nl   = lane & 15;
  const int kb8  = hi << 3;
  const int ksb  = hi << 4;

  __shared__ __bf16 Ks[FA_BK][DQK + 16];   // [key][d], row = 416B (16B aligned)
  __shared__ __bf16 Vst[DV_][FA_BK + 8];   // [vd][key]
  __shared__ __bf16 Ps[4][16][FA_BK + 8];  // per-wave P staging

  // ---- resident Q fragments for this wave's 16 rows (6 frags over d=192) ----
  const __bf16* Qb = Q + ((size_t)bh * S_ + q0 + wave * 16) * DQK;
  v16bf qf[6];
#pragma unroll
  for (int f = 0; f < 6; ++f) {
    const __bf16* qrow = Qb + (size_t)nl * DQK + f * 32;
#pragma unroll
    for (int v = 0; v < 8; ++v) {
      int kk = (v < 4) ? (kb8 + 2 * v) : (16 + kb8 + 2 * (v - 4));
      qf[f][2 * v]     = qrow[kk];
      qf[f][2 * v + 1] = qrow[kk + 1];
    }
  }

  const v8f vzero = {0.f,0.f,0.f,0.f,0.f,0.f,0.f,0.f};
  v8f oacc[8];
#pragma unroll
  for (int t = 0; t < 8; ++t) oacc[t] = vzero;
  float m_i[8], l_i[8];
#pragma unroll
  for (int r = 0; r < 8; ++r) { m_i[r] = -1e30f; l_i[r] = 0.f; }

  const float scale = 0.07216878364870323f;   // (DN+DR)^-0.5
  const int ktiles = blockIdx.x + 1;          // causal: FA_BK == FA_BM

  auto ldK = [&](int nc, int f) {
    v16bf kf;
    int ks = f * 32 + kb8;
#pragma unroll
    for (int i = 0; i < 16; ++i) kf[i] = Ks[nc][ks + (i & 7) + ((i >> 3) << 4)];
    return kf;
  };
  auto ldV = [&](int t, int f) {
    v16bf vf;
    int nc = t * 16 + nl;
    int ks = f * 32 + kb8;
#pragma unroll
    for (int i = 0; i < 16; ++i) vf[i] = Vst[nc][ks + (i & 7) + ((i >> 3) << 4)];
    return vf;
  };

  for (int kt = 0; kt < ktiles; ++kt) {
    const int k0 = kt * FA_BK;
    __syncthreads();
    // ---- stage K tile (b128 copies) ----
    {
      const uint4* src = (const uint4*)(Kp + ((size_t)bh * S_ + k0) * DQK);
      for (int i = tid; i < FA_BK * (DQK / 8); i += 128) {
        int row = i / 24, seg = i % 24;             // 24 uint4 per 192-elem row
        *(uint4*)&Ks[row][seg * 8] = src[row * 24 + seg];
      }
      // ---- stage V tile transposed: b128 loads + b16 scatter ----
      const uint4* vsrc = (const uint4*)(Vp + ((size_t)bh * S_ + k0) * DV_);
      for (int i = tid; i < FA_BK * (DV_ / 8); i += 128) {
        int row = i >> 4, seg = i & 15;
        uint4 v = vsrc[row * 16 + seg];
        const __bf16* pv = (const __bf16*)&v;
#pragma unroll
        for (int j = 0; j < 8; ++j) Vst[seg * 8 + j][row] = pv[j];
      }
    }
    __syncthreads();

    // ---- S = Q @ K^T (4 n-tiles x 6 k-steps, double-buffered K frags) ----
    v8f sacc[4];
#pragma unroll
    for (int t = 0; t < 4; ++t) {
      v8f s = vzero;
      int nc = t * 16 + nl;
      v16bf kcur = ldK(nc, 0);
#pragma unroll
      for (int f = 0; f < 6; ++f) {
        v16bf knx;
        if (f + 1 < 6) knx = ldK(nc, f + 1);
        s = __builtin_amdgcn_wmma_f32_16x16x32_bf16(
            false, qf[f], false, kcur, (short)0, s, false, false);
        if (f + 1 < 6) kcur = knx;
      }
      sacc[t] = s;
    }

    // ---- scale + causal mask ----
#pragma unroll
    for (int t = 0; t < 4; ++t)
#pragma unroll
      for (int r = 0; r < 8; ++r) {
        int qg = q0 + wave * 16 + r + 8 * hi;
        int kg = k0 + t * 16 + nl;
        float v = sacc[t][r] * scale;
        sacc[t][r] = (kg <= qg) ? v : -1e30f;
      }

    // ---- online softmax (row stats across the 16 lanes of a half-wave) ----
    float corrf[8];
#pragma unroll
    for (int r = 0; r < 8; ++r) {
      float mx = fmaxf(fmaxf(sacc[0][r], sacc[1][r]),
                       fmaxf(sacc[2][r], sacc[3][r]));
#pragma unroll
      for (int off = 1; off < 16; off <<= 1) mx = fmaxf(mx, __shfl_xor(mx, off, 16));
      float mnew = fmaxf(m_i[r], mx);
      float corr = __expf(m_i[r] - mnew);
      m_i[r] = mnew;
      float lsum = 0.f;
#pragma unroll
      for (int t = 0; t < 4; ++t) {
        float p = __expf(sacc[t][r] - mnew);
        sacc[t][r] = p;
        lsum += p;
      }
#pragma unroll
      for (int off = 1; off < 16; off <<= 1) lsum += __shfl_xor(lsum, off, 16);
      l_i[r] = l_i[r] * corr + lsum;
      corrf[r] = corr;
    }
#pragma unroll
    for (int t = 0; t < 8; ++t)
#pragma unroll
      for (int r = 0; r < 8; ++r) oacc[t][r] *= corrf[r];

    // ---- reshape P (C-layout) -> A-fragments via per-wave LDS stage ----
#pragma unroll
    for (int t = 0; t < 4; ++t)
#pragma unroll
      for (int r = 0; r < 8; ++r)
        Ps[wave][r + 8 * hi][t * 16 + nl] = (__bf16)sacc[t][r];
    __builtin_amdgcn_wave_barrier();   // keep stores before the reads below

    v16bf pf[2];
#pragma unroll
    for (int f = 0; f < 2; ++f) {
#pragma unroll
      for (int v = 0; v < 8; ++v) {
        int kk = ((v < 4) ? (kb8 + 2 * v) : (16 + kb8 + 2 * (v - 4))) + f * 32;
        pf[f][2 * v]     = Ps[wave][nl][kk];
        pf[f][2 * v + 1] = Ps[wave][nl][kk + 1];
      }
    }
    // ---- O += P @ V (16 steps flat, double-buffered V frags) ----
    v16bf vcur = ldV(0, 0);
#pragma unroll
    for (int idx = 0; idx < 16; ++idx) {
      int t = idx >> 1, f = idx & 1;
      v16bf vnx;
      if (idx + 1 < 16) vnx = ldV((idx + 1) >> 1, (idx + 1) & 1);
      oacc[t] = __builtin_amdgcn_wmma_f32_16x16x32_bf16(
          false, pf[f], false, vcur, (short)0, oacc[t], false, false);
      if (idx + 1 < 16) vcur = vnx;
    }
  }

  // ---- finalize: divide by l, write [B,S,H*DV] ----
#pragma unroll
  for (int r = 0; r < 8; ++r) {
    float inv = 1.f / l_i[r];
    int srow = q0 + wave * 16 + r + 8 * hi;
    size_t base = ((size_t)b * S_ + srow) * (H_ * DV_) + h * DV_;
#pragma unroll
    for (int t = 0; t < 8; ++t)
      O[base + t * 16 + nl] = oacc[t][r] * inv;
  }
}

// ---------------------------------------------------------------------------
// Host-side orchestration.
// ---------------------------------------------------------------------------
extern "C" void kernel_launch(void* const* d_in, const int* in_sizes, int n_in,
                              void* d_out, int out_size, void* d_ws, size_t ws_size,
                              hipStream_t stream) {
  (void)in_sizes; (void)n_in; (void)out_size; (void)ws_size;
  const float* x         = (const float*)d_in[0];
  /* d_in[1] = attention_mask (pure causal; applied analytically) */
  const float* w_q_down  = (const float*)d_in[2];
  const float* q_norm_g  = (const float*)d_in[3];
  const float* w_q_up    = (const float*)d_in[4];
  const float* w_q_rope  = (const float*)d_in[5];
  const float* w_kv_down = (const float*)d_in[6];
  const float* kv_norm_g = (const float*)d_in[7];
  const float* w_k_up    = (const float*)d_in[8];
  const float* w_v_up    = (const float*)d_in[9];
  const float* w_k_rope  = (const float*)d_in[10];
  const float* w_o       = (const float*)d_in[11];
  float* out = (float*)d_out;

  char* p = (char*)d_ws;
  float*  cq    = (float*)p;  p += (size_t)MTOT * CQ_ * 4;         // 25.2 MB
  float*  ckv   = (float*)p;  p += (size_t)MTOT * CKV_ * 4;        //  8.4 MB
  float*  krope = (float*)p;  p += (size_t)MTOT * DR_ * 4;         //  1.0 MB
  float*  big0  = (float*)p;  p += (size_t)MTOT * (H_ * DN_) * 4;  // 33.6 MB
  float*  big1  = (float*)p;  p += (size_t)MTOT * (H_ * DV_) * 4;  // 33.6 MB
  __bf16* qpack = (__bf16*)p; p += (size_t)B_ * H_ * S_ * DQK * 2; // 25.2 MB
  __bf16* kpack = (__bf16*)p; p += (size_t)B_ * H_ * S_ * DQK * 2; // 25.2 MB
  __bf16* vpack = (__bf16*)p; p += (size_t)B_ * H_ * S_ * DV_ * 2; // 16.8 MB

  // --- query path ---
  gemm_bf16_wmma<CQ_, D_><<<dim3(MTOT/GBM, CQ_/128), 256, 0, stream>>>(x, w_q_down, cq);
  rmsnorm_inplace<<<MTOT, 256, 0, stream>>>(cq, q_norm_g, CQ_);
  gemm_bf16_wmma<H_*DN_, CQ_><<<dim3(MTOT/GBM, (H_*DN_)/128), 256, 0, stream>>>(cq, w_q_up, big0);
  gemm_bf16_wmma<H_*DR_, CQ_><<<dim3(MTOT/GBM, (H_*DR_)/128), 256, 0, stream>>>(cq, w_q_rope, big1);
  // --- kv path ---
  gemm_bf16_wmma<CKV_, D_><<<dim3(MTOT/GBM, CKV_/128), 256, 0, stream>>>(x, w_kv_down, ckv);
  rmsnorm_inplace<<<MTOT, 256, 0, stream>>>(ckv, kv_norm_g, CKV_);
  gemm_bf16_wmma<DR_, D_><<<dim3(MTOT/GBM, 1), 256, 0, stream>>>(x, w_k_rope, krope);
  // --- pack q (rope fused), then k_nope / v, then pack k/v ---
  pack_q<<<(B_*H_*S_*DQK)/256, 256, 0, stream>>>(big0, big1, qpack);
  gemm_bf16_wmma<H_*DN_, CKV_><<<dim3(MTOT/GBM, (H_*DN_)/128), 256, 0, stream>>>(ckv, w_k_up, big0);
  gemm_bf16_wmma<H_*DV_, CKV_><<<dim3(MTOT/GBM, (H_*DV_)/128), 256, 0, stream>>>(ckv, w_v_up, big1);
  pack_k<<<(B_*H_*S_*DQK)/256, 256, 0, stream>>>(big0, krope, kpack);
  pack_v<<<(B_*H_*S_*DV_)/256, 256, 0, stream>>>(big1, vpack);
  // --- flash attention (writes [B,S,H*DV] into big0) ---
  fa_kernel<<<dim3(S_/FA_BM, B_*H_), 128, 0, stream>>>(qpack, kpack, vpack, big0);
  // --- output projection ---
  gemm_bf16_wmma<D_, H_*DV_><<<dim3(MTOT/GBM, D_/128), 256, 0, stream>>>(big0, w_o, out);
}
